// old_school_att_head_56796647522702
// MI455X (gfx1250) — matrix operand order repaired and checked by reading, not compiled, not run on hardware
//
#include <hip/hip_runtime.h>
#include <hip/hip_bf16.h>

typedef __attribute__((ext_vector_type(16))) __bf16 v16bf;
typedef __attribute__((ext_vector_type(8)))  float  v8f;

__device__ __forceinline__ unsigned short f2bf(float f) {
  union { float f; unsigned int u; } v; v.f = f;
  unsigned int r = v.u + 0x7FFFu + ((v.u >> 16) & 1u);   // round-to-nearest-even
  return (unsigned short)(r >> 16);
}

__global__ void f32_to_bf16_kernel(const float* __restrict__ in,
                                   unsigned short* __restrict__ out, long n) {
  long i = (long)blockIdx.x * blockDim.x + threadIdx.x;
  if (i < n) out[i] = f2bf(in[i]);
}

// Tiled convert: x (B,D,L) f32 -> xb (B,D,L) bf16 AND xbt (B,L,D) bf16 (transposed).
// 64x64 tile per block; all global traffic is 16B/32B coalesced.
__global__ __launch_bounds__(256) void cvt_transpose_kernel(
    const float* __restrict__ x, unsigned short* __restrict__ xb,
    unsigned short* __restrict__ xbt, int D, int L) {
  const int b  = blockIdx.z;
  const int l0 = blockIdx.x * 64;
  const int d0 = blockIdx.y * 64;
  __shared__ unsigned short sT[64 * 72];  // [d][l], pad 8 -> 144B rows (16B aligned)
  const int tid = threadIdx.x;
  const float* xg = x + (long)b * D * L;
  {
    const int r = tid >> 2;          // d offset 0..63
    const int c = (tid & 3) * 16;    // l offset 0,16,32,48
    const float4* src = (const float4*)(xg + (long)(d0 + r) * L + l0 + c);
    unsigned int pk[8];
#pragma unroll
    for (int i = 0; i < 4; ++i) {
      float4 f = src[i];
      pk[2 * i]     = (unsigned)f2bf(f.x) | ((unsigned)f2bf(f.y) << 16);
      pk[2 * i + 1] = (unsigned)f2bf(f.z) | ((unsigned)f2bf(f.w) << 16);
    }
    unsigned short* dst = xb + (long)b * D * L + (long)(d0 + r) * L + l0 + c;
    *(uint4*)dst       = *(uint4*)&pk[0];
    *(uint4*)(dst + 8) = *(uint4*)&pk[4];
    *(uint4*)(&sT[r * 72 + c])     = *(uint4*)&pk[0];
    *(uint4*)(&sT[r * 72 + c + 8]) = *(uint4*)&pk[4];
  }
  __syncthreads();
  {
    const int r = tid >> 2;          // l offset 0..63
    const int c = (tid & 3) * 16;    // d offset
    unsigned int pk[8];
#pragma unroll
    for (int i = 0; i < 8; ++i) {
      unsigned short lo = sT[(c + 2 * i) * 72 + r];
      unsigned short hi = sT[(c + 2 * i + 1) * 72 + r];
      pk[i] = (unsigned)lo | ((unsigned)hi << 16);
    }
    unsigned short* dst = xbt + (long)b * L * D + (long)(l0 + r) * D + d0 + c;
    *(uint4*)dst       = *(uint4*)&pk[0];
    *(uint4*)(dst + 8) = *(uint4*)&pk[4];
  }
}

// bf16 GEMM: C = alpha * A(MxK, row-major, stride K) * B(KxN) where B is supplied
// N-MAJOR: Bn[n][k], stride K. Both LDS panels stage as straight float4 copies.
// Double-buffered LDS, one barrier per K-step. 8 waves, 128x128 tile, BK=32,
// wave computes 64x32 (4x2 WMMA tiles).
// MODE 0: store bf16 row-major to Cb.
// MODE 1: store f32 Cf; skip macro tiles fully below causal diagonal.
// MODE 2: store f32 Cf; clip K loop to n0+128 (Bn rows beyond are zero).
template <int MODE>
__global__ __launch_bounds__(256) void gemm_bf16_kernel(
    const unsigned short* __restrict__ A, const unsigned short* __restrict__ Bn,
    float* __restrict__ Cf, unsigned short* __restrict__ Cb,
    int M, int N, int K, long strideA, long strideB, long strideC, float alpha) {
  const int b  = blockIdx.z;
  const int n0 = blockIdx.x * 128;
  const int m0 = blockIdx.y * 128;
  if (MODE == 1 && m0 > n0 + 127) return;  // fully masked: softmax never reads it

  A  += (long)b * strideA;
  Bn += (long)b * strideB;

  int kTiles = K / 32;
  if (MODE == 2) { int kmax = n0 + 128; if (kmax < K) kTiles = kmax / 32; }

  const int tid  = threadIdx.x;
  const int lane = tid & 31;
  const int wave = tid >> 5;
  const int wm   = wave & 1;   // 2 waves along M
  const int wn   = wave >> 1;  // 4 waves along N

  __shared__ unsigned short sA[2][128 * 40];  // [m][k], 80B rows (16B aligned)
  __shared__ unsigned short sB[2][128 * 40];  // [n][k]

  v8f acc[4][2] = {};

  const int frow = lane & 15;
  const int ak0  = (lane >> 4) * 8;
  const int bk0  = (lane >> 4) * 8;

  const int r = tid >> 2;            // 0..63
  const int c = (tid & 3) * 8;       // 0,8,16,24
  const unsigned short* pa0 = A  + (long)(m0 + r) * K + c;
  const unsigned short* pa1 = A  + (long)(m0 + r + 64) * K + c;
  const unsigned short* pb0 = Bn + (long)(n0 + r) * K + c;
  const unsigned short* pb1 = Bn + (long)(n0 + r + 64) * K + c;

  float4 ra0 = *(const float4*)pa0, ra1 = *(const float4*)pa1;
  float4 rb0 = *(const float4*)pb0, rb1 = *(const float4*)pb1;
  // prologue: stage tile 0 into buffer 0
  *(float4*)(&sA[0][r * 40 + c]) = ra0;  *(float4*)(&sA[0][(r + 64) * 40 + c]) = ra1;
  *(float4*)(&sB[0][r * 40 + c]) = rb0;  *(float4*)(&sB[0][(r + 64) * 40 + c]) = rb1;

  for (int kt = 0; kt < kTiles; ++kt) {
    const int  cur  = kt & 1;
    const bool more = (kt + 1 < kTiles);
    if (more) {                       // global -> regs for tile kt+1
      const long off = (long)(kt + 1) * 32;
      ra0 = *(const float4*)(pa0 + off);  ra1 = *(const float4*)(pa1 + off);
      rb0 = *(const float4*)(pb0 + off);  rb1 = *(const float4*)(pb1 + off);
    }
    __syncthreads();                  // s_wait_dscnt 0 + barrier: orders LDS r/w
    if (more) {                       // regs -> LDS alternate buffer
      const int nxt = cur ^ 1;
      *(float4*)(&sA[nxt][r * 40 + c]) = ra0;  *(float4*)(&sA[nxt][(r + 64) * 40 + c]) = ra1;
      *(float4*)(&sB[nxt][r * 40 + c]) = rb0;  *(float4*)(&sB[nxt][(r + 64) * 40 + c]) = rb1;
    }
    union Frag { v16bf v; float4 f[2]; };
    Frag af[4], bfr[2];
#pragma unroll
    for (int im = 0; im < 4; ++im) {  // elems 0..7 = K[ak0..+7], 8..15 = K[ak0+16..+23]
      const int m = wm * 64 + im * 16 + frow;
      af[im].f[0] = *(const float4*)(&sA[cur][m * 40 + ak0]);
      af[im].f[1] = *(const float4*)(&sA[cur][m * 40 + ak0 + 16]);
    }
#pragma unroll
    for (int in = 0; in < 2; ++in) {  // elems 0..7 = K[bk0..+7], 8..15 = K[bk0+16..+23]
      const int n = wn * 32 + in * 16 + frow;
      bfr[in].f[0] = *(const float4*)(&sB[cur][n * 40 + bk0]);
      bfr[in].f[1] = *(const float4*)(&sB[cur][n * 40 + bk0 + 16]);
    }
#pragma unroll
    for (int im = 0; im < 4; ++im)
#pragma unroll
      for (int in = 0; in < 2; ++in)
        acc[im][in] = __builtin_amdgcn_wmma_f32_16x16x32_bf16(
            false, af[im].v, false, bfr[in].v, (short)0, acc[im][in], false, false);
  }

  const int cn  = lane & 15;
  const int cm8 = (lane >> 4) * 8;
#pragma unroll
  for (int im = 0; im < 4; ++im)
#pragma unroll
    for (int in = 0; in < 2; ++in) {
      const int mb = m0 + wm * 64 + im * 16 + cm8;
      const int nn = n0 + wn * 32 + in * 16 + cn;
#pragma unroll
      for (int e = 0; e < 8; ++e) {
        const float v = acc[im][in][e] * alpha;
        const int m = mb + e;
        if constexpr (MODE == 0)
          Cb[(long)b * strideC + (long)m * N + nn] = f2bf(v);
        else
          Cf[(long)b * strideC + (long)m * N + nn] = v;
      }
    }
}

// Column softmax with causal mask; writes P TRANSPOSED (Pt[m][l]) so GEMM3's
// B operand is N-major. Thread m owns row m of Pt -> contiguous 16B stores.
__global__ __launch_bounds__(256) void colsoftmax_kernel(
    const float* __restrict__ corr, unsigned short* __restrict__ Pt, int L) {
  const int b = blockIdx.y;
  const int m = blockIdx.x * 256 + threadIdx.x;
  const float* c = corr + (long)b * L * L + m;
  unsigned short* pr = Pt + (long)b * L * L + (long)m * L;
  float mx = -3.402823466e38f, sum = 0.f;
  for (int l = 0; l <= m; ++l) {        // online max/sum (reads coalesced across lanes)
    const float v  = c[(long)l * L];
    const float nm = fmaxf(mx, v);
    sum = sum * __expf(mx - nm) + __expf(v - nm);
    mx  = nm;
  }
  const float inv = 1.f / sum;
  const int mp1  = m + 1;
  const int full = mp1 & ~7;
  for (int l8 = 0; l8 < full; l8 += 8) {
    unsigned int pk[4];
#pragma unroll
    for (int i = 0; i < 4; ++i) {
      const unsigned short lo = f2bf(__expf(c[(long)(l8 + 2 * i) * L] - mx) * inv);
      const unsigned short hi = f2bf(__expf(c[(long)(l8 + 2 * i + 1) * L] - mx) * inv);
      pk[i] = (unsigned)lo | ((unsigned)hi << 16);
    }
    *(uint4*)(pr + l8) = *(uint4*)pk;
  }
  for (int l = full; l <= m; ++l)
    pr[l] = f2bf(__expf(c[(long)l * L] - mx) * inv);
  const int zalign = (mp1 + 7) & ~7;
  for (int l = mp1; l < zalign && l < L; ++l) pr[l] = 0;
  const uint4 z = {0, 0, 0, 0};
  for (int l8 = zalign; l8 < L; l8 += 8) *(uint4*)(pr + l8) = z;
}

extern "C" void kernel_launch(void* const* d_in, const int* in_sizes, int n_in,
                              void* d_out, int out_size, void* d_ws, size_t ws_size,
                              hipStream_t stream) {
  const float* x = (const float*)d_in[0];  // (8, 1024, 2048)
  const float* W = (const float*)d_in[1];  // (1024, 1024)
  const int B = 8, D = 1024, L = 2048;

  char* ws = (char*)d_ws;
  unsigned short* Wb  = (unsigned short*)ws; ws += (size_t)D * D * 2;       //  2 MB
  unsigned short* xb  = (unsigned short*)ws; ws += (size_t)B * D * L * 2;   // 32 MB  x   (D,L) bf16
  unsigned short* xbt = (unsigned short*)ws; ws += (size_t)B * L * D * 2;   // 32 MB  x^T (L,D) bf16
  unsigned short* xpt = (unsigned short*)ws; ws += (size_t)B * L * D * 2;   // 32 MB  x_^T (L,D) bf16
  float*          corr = (float*)ws;         ws += (size_t)B * L * L * 4;   // 128 MB
  unsigned short* Pt  = (unsigned short*)ws;                                 // 64 MB  p^T (L,L) bf16

  f32_to_bf16_kernel<<<(D * D + 255) / 256, 256, 0, stream>>>(W, Wb, (long)D * D);
  cvt_transpose_kernel<<<dim3(L / 64, D / 64, B), 256, 0, stream>>>(x, xb, xbt, D, L);

  // GEMM1': x_^T = x^T @ W^T  (M=L, N=D, K=D); N-major of W^T is W row-major.
  gemm_bf16_kernel<0><<<dim3(D / 128, L / 128, B), 256, 0, stream>>>(
      xbt, Wb, nullptr, xpt, L, D, D, (long)L * D, 0L, (long)L * D, 1.0f);

  // GEMM2: corr = (1/32) * x_^T @ x_  (M=L, N=L, K=D); A and Bn both from xpt.
  gemm_bf16_kernel<1><<<dim3(L / 128, L / 128, B), 256, 0, stream>>>(
      xpt, xpt, corr, nullptr, L, L, D, (long)L * D, (long)L * D, (long)L * L,
      0.03125f);

  colsoftmax_kernel<<<dim3(L / 256, B), 256, 0, stream>>>(corr, Pt, L);

  // GEMM3: out = x @ p  (M=D, N=L, K=L); Bn = p^T; K clipped via zeros in p^T.
  gemm_bf16_kernel<2><<<dim3(L / 128, D / 128, B), 256, 0, stream>>>(
      xb, Pt, (float*)d_out, nullptr, D, L, L, (long)D * L, (long)L * L,
      (long)D * L, 1.0f);
}
